// CAModel_31001073942585
// MI455X (gfx1250) — matrix operand order repaired and checked by reading, compiled, not run on hardware
//
#include <hip/hip_runtime.h>

// ---------------------------------------------------------------------------
// Fused NCA step for MI455X (gfx1250, wave32, WMMA).
// Both per-pixel matmuls run on v_wmma_f32_16x16x32_bf16 (f32 accumulate).
// One workgroup (256 thr = 8 waves) computes a 16x16 MLP region and writes
// the interior 14x14 pixels (halo recompute makes the post-update alpha
// max-pool block-local, so the whole step is ONE kernel: ~276MB HBM traffic
// -> ~12us floor @ 23.3TB/s; bf16 matrix ops (~34 GFLOP) stay under that).
// ---------------------------------------------------------------------------

typedef __attribute__((ext_vector_type(16))) __bf16 v16bf;
typedef __attribute__((ext_vector_type(8)))  __bf16 v8bf;
typedef __attribute__((ext_vector_type(8)))  float  v8f;

#define NCA_H 256
#define NCA_W 256
#define NCA_C 16
#define NCA_HID 128
#define OUT_TILE 14              // interior written per block
#define NTILE 19                 // ceil(256/14)

// LDS layout (bytes, all 16B aligned)
#define XS_OFF    0              // f32 xs[16][18][18]          20736 B
#define W1_OFF    20736          // bf16 w1b[128][72] (K pad64) 18432 B
#define W2_OFF    39168          // bf16 w2b[16][136]            4352 B
#define B1_OFF    43520          // f32 b1s[128]                  512 B
#define B2_OFF    44032          // f32 b2s[16]                    64 B
#define FT_OFF    44096          // bf16 featb[256][72]         36864 B
#define HB_OFF    80960          // bf16 hb[256][136]           69632 B
#define AN_OFF    150592         // f32 an[256] (new alpha)      1024 B
#define SMEM_BYTES 151616

#define FSTR 72                  // featb row stride (bf16 elems)
#define WSTR 72                  // w1b row stride
#define HSTR 136                 // hb / w2b row stride

__device__ __forceinline__ int wrapc(int v, int n) {
    v += (v < 0) ? n : 0;
    v -= (v >= n) ? n : 0;
    return v;
}

__device__ __forceinline__ v8f wmma_bf16(v16bf a, v16bf b, v8f c) {
    // D = A(16x32 bf16) * B(32x16 bf16) + C(16x16 f32)
    return __builtin_amdgcn_wmma_f32_16x16x32_bf16(
        /*neg_a=*/false, a, /*neg_b=*/false, b,
        /*c_mod=*/(short)0, c, /*reuse_a=*/false, /*reuse_b=*/false);
}

// A-fragment (16x32 bf16) in contiguous K order, not the ISA table order: lane holds row M=lane&15;
// lanes 0-15: K = k0+{0..15}; lanes 16-31: K = k0+{16..31} (matches loadB).
__device__ __forceinline__ v16bf loadA(const __bf16* wlds, int stride,
                                       int m0, int k0, int lane) {
    const int m    = m0 + (lane & 15);
    const int koff = k0 + ((lane & 16) ? 16 : 0);
    const __bf16* base = wlds + m * stride + koff;
    v8bf lo = *(const v8bf*)(base);        // K run 0..7
    v8bf hi = *(const v8bf*)(base + 8);    // K run 8..15
    return __builtin_shufflevector(lo, hi, 0, 1, 2, 3, 4, 5, 6, 7,
                                           8, 9, 10, 11, 12, 13, 14, 15);
}

// B-fragment (32x16 bf16): lane = column N (&15); lanes 0-15 hold K k0..k0+15,
// lanes 16-31 hold K k0+16..k0+31; contiguous 32B per lane from row-major LDS.
__device__ __forceinline__ v16bf loadB(const __bf16* flds, int stride,
                                       int pix, int k0, int lane) {
    const int koff = k0 + ((lane & 16) ? 16 : 0);
    return *(const v16bf*)(flds + pix * stride + koff);
}

__global__ void __launch_bounds__(256)
nca_step_kernel(const float* __restrict__ x,  const float* __restrict__ w1,
                const float* __restrict__ b1, const float* __restrict__ w2,
                const float* __restrict__ b2, const int* __restrict__ fire,
                float* __restrict__ out) {
    extern __shared__ char smem[];
    float*  xs    = (float*)(smem + XS_OFF);
    __bf16* w1b   = (__bf16*)(smem + W1_OFF);
    __bf16* w2b   = (__bf16*)(smem + W2_OFF);
    float*  b1s   = (float*)(smem + B1_OFF);
    float*  b2s   = (float*)(smem + B2_OFF);
    __bf16* featb = (__bf16*)(smem + FT_OFF);
    __bf16* hb    = (__bf16*)(smem + HB_OFF);
    float*  an    = (float*)(smem + AN_OFF);

    const int tid  = threadIdx.x;
    const int lane = tid & 31;
    const int wave = tid >> 5;
    const int bimg = blockIdx.z;
    const int oy   = blockIdx.y * OUT_TILE;   // global row of interior origin
    const int ox   = blockIdx.x * OUT_TILE;

    // ---- stage weights / biases (bf16, K padded 48->64 with zeros) ----
    for (int i = tid; i < NCA_HID * 64; i += 256) {
        const int o = i >> 6, k = i & 63;
        w1b[o * WSTR + k] = (k < 48) ? (__bf16)w1[o * 48 + k] : (__bf16)0.0f;
    }
    for (int i = tid; i < NCA_C * NCA_HID; i += 256) {
        const int o = i >> 7, k = i & 127;
        w2b[o * HSTR + k] = (__bf16)w2[o * NCA_HID + k];
    }
    if (tid < NCA_HID) b1s[tid] = b1[tid];
    if (tid < NCA_C)   b2s[tid] = b2[tid];

    // ---- stage x tile [16ch][18][18] with wrap ----
    const float* xb = x + (size_t)bimg * NCA_C * NCA_H * NCA_W;
    for (int i = tid; i < NCA_C * 324; i += 256) {
        const int c = i / 324;
        const int rc = i - c * 324;
        const int r = rc / 18, col = rc - r * 18;
        const int gy = wrapc(oy + r - 1, NCA_H);
        const int gx = wrapc(ox + col - 1, NCA_W);
        xs[i] = xb[(size_t)c * NCA_H * NCA_W + gy * NCA_W + gx];
    }
    __syncthreads();

    // ---- perception: 1 thread = 1 pixel, 48 bf16 features into B-layout ----
    {
        const int ty = tid >> 4, tx = tid & 15;
        __bf16* f = featb + tid * FSTR;
        #pragma unroll
        for (int c = 0; c < NCA_C; ++c) {
            const float* p0 = xs + c * 324 + ty * 18 + tx;
            const float n00 = p0[0],  n01 = p0[1],  n02 = p0[2];
            const float n10 = p0[18], n11 = p0[19], n12 = p0[20];
            const float n20 = p0[36], n21 = p0[37], n22 = p0[38];
            const float sx = 0.125f * (n02 - n00 + n22 - n20) + 0.25f * (n12 - n10);
            const float sy = 0.125f * (n20 - n00 + n22 - n02) + 0.25f * (n21 - n01);
            f[3 * c + 0] = (__bf16)n11;   // identity
            f[3 * c + 1] = (__bf16)sx;    // dx
            f[3 * c + 2] = (__bf16)sy;    // dy
        }
        #pragma unroll
        for (int k = 48; k < 64; ++k) f[k] = (__bf16)0.0f;
    }
    __syncthreads();

    // ---- MLP: each wave owns pixel rows g = 2*wave, 2*wave+1 ----
    const int col  = lane & 15;
    const int half = lane >> 4;
    const int cb   = half * 8;           // D-fragment M base (also channel base)
    float xn[2][8];

    #pragma unroll
    for (int gi = 0; gi < 2; ++gi) {
        const int g = 2 * wave + gi;     // pixel row in tile
        const int p = g * 16 + col;      // this lane's B-column pixel

        // layer-1 B fragments (K=0..63, features padded) — reused for 8 tiles
        const v16bf B0 = loadB(featb, FSTR, p, 0,  lane);
        const v16bf B1 = loadB(featb, FSTR, p, 32, lane);

        // layer 1: 8 output tiles x (2 chained WMMAs)
        #pragma unroll
        for (int m0 = 0; m0 < NCA_HID; m0 += 16) {
            v8f acc = {0.f, 0.f, 0.f, 0.f, 0.f, 0.f, 0.f, 0.f};
            acc = wmma_bf16(loadA(w1b, WSTR, m0, 0,  lane), B0, acc);
            acc = wmma_bf16(loadA(w1b, WSTR, m0, 32, lane), B1, acc);
            v8bf hc;
            #pragma unroll
            for (int v = 0; v < 8; ++v) {
                float hv = acc[v] + b1s[m0 + cb + v];
                hc[v] = (__bf16)fmaxf(hv, 0.0f);   // ReLU
            }
            // D layout: VGPR v -> M = cb+v, N = col  => contiguous channels
            *(v8bf*)(hb + p * HSTR + m0 + cb) = hc;
        }

        // layer 2: 16 outputs, K=128 -> 4 chained WMMAs
        v8f acc2 = {0.f, 0.f, 0.f, 0.f, 0.f, 0.f, 0.f, 0.f};
        #pragma unroll
        for (int k0 = 0; k0 < NCA_HID; k0 += 32) {
            acc2 = wmma_bf16(loadA(w2b, HSTR, 0, k0, lane),
                             loadB(hb, HSTR, p, k0, lane), acc2);
        }

        // residual update (f32), fire mask
        const int gy = wrapc(oy + g - 1, NCA_H);
        const int gx = wrapc(ox + col - 1, NCA_W);
        const float fm = (float)fire[(size_t)bimg * NCA_H * NCA_W + gy * NCA_W + gx];
        #pragma unroll
        for (int v = 0; v < 8; ++v) {
            const int c = cb + v;
            const float xv = xs[c * 324 + (g + 1) * 18 + (col + 1)];
            xn[gi][v] = xv + (acc2[v] + b2s[c]) * fm;   // STEP_SIZE = 1
        }
        if (half == 0) an[p] = xn[gi][3];  // new alpha (channel 3 sits in v=3)
    }
    __syncthreads();

    // ---- alive masks + interior store ----
    #pragma unroll
    for (int gi = 0; gi < 2; ++gi) {
        const int g = 2 * wave + gi;
        if (g >= 1 && g <= OUT_TILE && col >= 1 && col <= OUT_TILE) {
            const int gy = oy + g - 1, gx = ox + col - 1;
            if (gy < NCA_H && gx < NCA_W) {
                // pre-life: 3x3 max of original alpha (xs channel 3)
                float mpre = -3.4e38f;
                #pragma unroll
                for (int dy = 0; dy < 3; ++dy)
                    #pragma unroll
                    for (int dx = 0; dx < 3; ++dx)
                        mpre = fmaxf(mpre, xs[3 * 324 + (g + dy) * 18 + (col + dx)]);
                // post-life: 3x3 max of updated alpha (block-local, interior)
                float mpost = -3.4e38f;
                #pragma unroll
                for (int dy = -1; dy <= 1; ++dy)
                    #pragma unroll
                    for (int dx = -1; dx <= 1; ++dx)
                        mpost = fmaxf(mpost, an[(g + dy) * 16 + (col + dx)]);
                const float life = (mpre > 0.1f && mpost > 0.1f) ? 1.0f : 0.0f;
                float* ob = out + (size_t)bimg * NCA_C * NCA_H * NCA_W;
                #pragma unroll
                for (int v = 0; v < 8; ++v)
                    ob[(size_t)(cb + v) * NCA_H * NCA_W + gy * NCA_W + gx] =
                        xn[gi][v] * life;
            }
        }
    }
}

extern "C" void kernel_launch(void* const* d_in, const int* in_sizes, int n_in,
                              void* d_out, int out_size, void* d_ws, size_t ws_size,
                              hipStream_t stream) {
    const float* x    = (const float*)d_in[0];
    const float* w1   = (const float*)d_in[1];
    const float* b1   = (const float*)d_in[2];
    const float* w2   = (const float*)d_in[3];
    const float* b2   = (const float*)d_in[4];
    const int*   fire = (const int*)d_in[5];
    float*       out  = (float*)d_out;

    (void)in_sizes; (void)n_in; (void)out_size; (void)d_ws; (void)ws_size;

    // allow >64KB dynamic LDS (gfx1250 WGP has 320KB); not a stream op,
    // safe under graph capture, idempotent.
    hipFuncSetAttribute((const void*)nca_step_kernel,
                        hipFuncAttributeMaxDynamicSharedMemorySize, SMEM_BYTES);

    dim3 grid(NTILE, NTILE, 32);
    nca_step_kernel<<<grid, 256, SMEM_BYTES, stream>>>(x, w1, b1, w2, b2, fire, out);
}
